// DEIMTransformerModule_33200097198137
// MI455X (gfx1250) — compile-verified, matrix operand order checked by reading, not run on hardware
//
#include <hip/hip_runtime.h>
#include <hip/hip_bf16.h>
#include <stdint.h>

#define B_     16
#define N_     300
#define D_     256
#define HEADS_ 8
#define HD_    32
#define M_     (B_*N_)    // 4800 tokens, divisible by 64
#define NPAD_  320        // keys padded to multiple of 32
#define TP_    16
#define NP_    4
#define L_     8500
#define HID_   512
#define FF2_   1024

#define BM_ 64
#define BN_ 64
#define BK_ 32

typedef __bf16 bf16;
typedef __bf16 v8bf  __attribute__((ext_vector_type(8)));
typedef __bf16 v16bf __attribute__((ext_vector_type(16)));
typedef float  v8f   __attribute__((ext_vector_type(8)));

#define WMMA_BF16(a, b, c) \
  __builtin_amdgcn_wmma_f32_16x16x32_bf16(false, (a), false, (b), (short)0, (c), false, false)

// ---------------------------------------------------------------------------
// bf16 WMMA fragment load, row-major source with leading dim `ld` (elements).
// CDNA5 16-bit A/B layout: lane = 16*half + m; half 0 holds K {0..7,16..23},
// half 1 holds K {8..15,24..31} for the 32-wide K step. Two B128 loads.
// Works for both global and LDS-backed (generic) pointers.
// ---------------------------------------------------------------------------
__device__ __forceinline__ v16bf frag_load(const bf16* __restrict__ base, int ld,
                                           int row0, int k0, int lane) {
  const int m    = lane & 15;
  const int half = lane >> 4;
  const bf16* p = base + (size_t)(row0 + m) * (size_t)ld + (size_t)(k0 + half * 8);
  v8bf lo = *(const v8bf*)(p);
  v8bf hi = *(const v8bf*)(p + 16);
  v16bf f;
#pragma unroll
  for (int i = 0; i < 8; ++i) { f[i] = lo[i]; f[i + 8] = hi[i]; }
  return f;
}

// Issue one async global->LDS B128 copy (tracked by ASYNCcnt).
__device__ __forceinline__ void async_b128(uint32_t lds_addr, const void* gaddr) {
  asm volatile("global_load_async_to_lds_b128 %0, %1, off"
               :: "v"(lds_addr), "v"(gaddr) : "memory");
}
__device__ __forceinline__ uint32_t lds_addr_of(const void* p) {
  return (uint32_t)(uintptr_t)p;  // low 32 bits of generic shared ptr = LDS offset
}

// ---------------------------------------------------------------------------
// Elementwise f32 -> bf16 convert
// ---------------------------------------------------------------------------
__global__ void k_cvt(const float* __restrict__ src, bf16* __restrict__ dst, int n) {
  int i = blockIdx.x * 256 + threadIdx.x;
  if (i < n) dst[i] = (bf16)src[i];
}

// ---------------------------------------------------------------------------
// C = A(MxK) x W(NoutxK)^T + bias.
// 8-wave workgroup computes a 64x64 super-tile; A/W K-slabs are staged into
// LDS with double-buffered async global->LDS B128 copies (ASYNCcnt pipeline),
// each wave runs two v_wmma_f32_16x16x32_bf16 per K step from LDS fragments.
// ---------------------------------------------------------------------------
__global__ __launch_bounds__(256)
void k_gemm(const bf16* __restrict__ A, const bf16* __restrict__ W,
            const float* __restrict__ bias, float* __restrict__ Cf,
            bf16* __restrict__ Cb, int M, int Nout, int K) {
  __shared__ __align__(16) bf16 sA[2][BM_ * BK_];
  __shared__ __align__(16) bf16 sW[2][BN_ * BK_];
  const int tid  = threadIdx.x;
  const int lane = tid & 31;
  const int wv   = tid >> 5;      // 0..7
  const int wm   = wv >> 1;       // 0..3  -> 16-row slab
  const int wn   = wv & 1;        // 0..1  -> 32-col slab
  const int tm = blockIdx.x, tn = blockIdx.y;

  // staging assignment: thread -> (row, 8-col chunk) = one B128 per tile
  const int lrow = tid >> 2;           // 0..63
  const int lc8  = (tid & 3) * 8;      // 0,8,16,24
  const bf16* gA = A + (size_t)(tm * BM_ + lrow) * K + lc8;
  const bf16* gW = W + (size_t)(tn * BN_ + lrow) * K + lc8;
  const uint32_t la[2] = { lds_addr_of(&sA[0][lrow * BK_ + lc8]),
                           lds_addr_of(&sA[1][lrow * BK_ + lc8]) };
  const uint32_t lw[2] = { lds_addr_of(&sW[0][lrow * BK_ + lc8]),
                           lds_addr_of(&sW[1][lrow * BK_ + lc8]) };

  v8f acc0 = {}, acc1 = {};
  const int steps = K / BK_;

  // prologue: stage K-step 0 into buffer 0
  async_b128(la[0], gA);
  async_b128(lw[0], gW);

  for (int i = 0; i < steps; ++i) {
    const int p = i & 1;
    __syncthreads();                    // buffer p^1 free for re-staging
    if (i + 1 < steps) {
      async_b128(la[p ^ 1], gA + (size_t)(i + 1) * BK_);
      async_b128(lw[p ^ 1], gW + (size_t)(i + 1) * BK_);
      asm volatile("s_wait_asynccnt 2" ::: "memory");  // step i landed, i+1 in flight
    } else {
      asm volatile("s_wait_asynccnt 0" ::: "memory");
    }
    __syncthreads();                    // buffer p populated by all waves

    v16bf a  = frag_load(sA[p], BK_, wm * 16, 0, lane);
    v16bf b0 = frag_load(sW[p], BK_, wn * 32, 0, lane);
    v16bf b1 = frag_load(sW[p], BK_, wn * 32 + 16, 0, lane);
    acc0 = WMMA_BF16(a, b0, acc0);
    acc1 = WMMA_BF16(a, b1, acc1);
  }

  const int rbase = tm * BM_ + wm * 16 + (lane >> 4) * 8;
#pragma unroll
  for (int j = 0; j < 2; ++j) {
    const int n = tn * BN_ + wn * 32 + j * 16 + (lane & 15);
    const float bv = bias ? bias[n] : 0.0f;
    const v8f acc = j ? acc1 : acc0;
#pragma unroll
    for (int r = 0; r < 8; ++r) {
      const int row = rbase + r;
      const float v = acc[r] + bv;
      if (Cf) Cf[(size_t)row * Nout + n] = v;
      if (Cb) Cb[(size_t)row * Nout + n] = (bf16)v;
    }
  }
}

// ---------------------------------------------------------------------------
// Repack qkv (M x 768 f32) into Q,K [bh][NPAD][32] bf16 and Vt [bh][32][NPAD].
// Padded rows zeroed so WMMA over the padded range is exact.
// ---------------------------------------------------------------------------
__global__ void k_repack(const float* __restrict__ qkv, bf16* __restrict__ Qb,
                         bf16* __restrict__ Kb, bf16* __restrict__ Vtb) {
  const int total = B_ * HEADS_ * NPAD_ * HD_;
  int i = blockIdx.x * 256 + threadIdx.x;
  if (i >= total) return;
  const int hd  = i & 31;
  const int row = (i >> 5) % NPAD_;
  const int bh  = i / (NPAD_ * HD_);
  const int b = bh >> 3, h = bh & 7;
  float q = 0.f, k = 0.f, v = 0.f;
  if (row < N_) {
    const size_t t = (size_t)(b * N_ + row) * 768;
    q = qkv[t + h * HD_ + hd];
    k = qkv[t + 256 + h * HD_ + hd];
    v = qkv[t + 512 + h * HD_ + hd];
  }
  Qb [(size_t)bh * NPAD_ * HD_ + (size_t)row * HD_ + hd] = (bf16)q;
  Kb [(size_t)bh * NPAD_ * HD_ + (size_t)row * HD_ + hd] = (bf16)k;
  Vtb[(size_t)bh * HD_ * NPAD_ + (size_t)hd * NPAD_ + row] = (bf16)v;
}

// ---------------------------------------------------------------------------
// Self-attention: one wave per (bh, q-tile). Scores via WMMA -> LDS strip,
// softmax in-wave, probs as bf16 in LDS, then probs x V via WMMA from LDS.
// ---------------------------------------------------------------------------
__global__ __launch_bounds__(32)
void k_attn(const bf16* __restrict__ Q, const bf16* __restrict__ Km,
            const bf16* __restrict__ Vt, float* __restrict__ SA) {
  __shared__ __align__(16) float sS[16 * NPAD_];
  __shared__ __align__(16) bf16  sP[16 * NPAD_];
  const int lane = threadIdx.x;
  const int bh = blockIdx.x;   // 0..127
  const int qt = blockIdx.y;   // 0..18 (rows 0..303; >=300 masked at store)
  const bf16* Qp = Q  + (size_t)bh * NPAD_ * HD_;
  const bf16* Kp = Km + (size_t)bh * NPAD_ * HD_;
  const bf16* Vp = Vt + (size_t)bh * HD_ * NPAD_;
  const float scale = 0.17677669529663687f;  // 1/sqrt(32)

  v16bf aq = frag_load(Qp, HD_, qt * 16, 0, lane);
#pragma unroll 4
  for (int kt = 0; kt < NPAD_ / 16; ++kt) {
    v16bf bk = frag_load(Kp, HD_, kt * 16, 0, lane);
    v8f acc = {};
    acc = WMMA_BF16(aq, bk, acc);
    const int n  = kt * 16 + (lane & 15);
    const int r0 = (lane >> 4) * 8;
#pragma unroll
    for (int r = 0; r < 8; ++r) sS[(r0 + r) * NPAD_ + n] = acc[r] * scale;
  }
  __syncthreads();

  if (lane < 16) {
    float mx = -3.4e38f;
    for (int c = 0; c < N_; ++c) mx = fmaxf(mx, sS[lane * NPAD_ + c]);
    float s = 0.f;
    for (int c = 0; c < N_; ++c) {
      const float e = __expf(sS[lane * NPAD_ + c] - mx);
      sS[lane * NPAD_ + c] = e;
      s += e;
    }
    const float inv = 1.0f / s;
    for (int c = 0; c < N_; ++c) sP[lane * NPAD_ + c] = (bf16)(sS[lane * NPAD_ + c] * inv);
    for (int c = N_; c < NPAD_; ++c) sP[lane * NPAD_ + c] = (bf16)0.0f;
  }
  __syncthreads();

  const int b = bh >> 3, h = bh & 7;
#pragma unroll
  for (int j = 0; j < 2; ++j) {  // two 16-wide tiles of HD=32
    v8f acc = {};
    for (int ks = 0; ks < NPAD_; ks += 32) {
      v16bf ap = frag_load(sP, NPAD_, 0, ks, lane);     // probs from LDS (ds loads)
      v16bf bv = frag_load(Vp, NPAD_, j * 16, ks, lane);
      acc = WMMA_BF16(ap, bv, acc);
    }
    const int hd = j * 16 + (lane & 15);
    const int r0 = (lane >> 4) * 8;
#pragma unroll
    for (int r = 0; r < 8; ++r) {
      const int qrow = qt * 16 + r0 + r;
      if (qrow < N_)
        SA[(size_t)(b * N_ + qrow) * D_ + h * HD_ + hd] = acc[r];
    }
  }
}

// ---------------------------------------------------------------------------
// Residual add + RMSNorm; one wave per 256-wide row, __shfl_xor reduction.
// Optional clip (final layer) and optional bf16 mirror for next GEMM.
// ---------------------------------------------------------------------------
__global__ __launch_bounds__(256)
void k_addnorm(const float* __restrict__ A, const float* __restrict__ Bm,
               const float* __restrict__ w, float* __restrict__ Of,
               bf16* __restrict__ Ob, int clip) {
  const int lane = threadIdx.x & 31;
  const int wid  = threadIdx.x >> 5;
  const int row  = blockIdx.x * 8 + wid;
  const float* a  = A  + (size_t)row * D_;
  const float* bm = Bm + (size_t)row * D_;
  float vals[8];
  float ss = 0.f;
#pragma unroll
  for (int i = 0; i < 8; ++i) {
    const int c = i * 32 + lane;
    float x = a[c] + bm[c];
    if (clip) x = fminf(fmaxf(x, -65504.0f), 65504.0f);
    vals[i] = x;
    ss += x * x;
  }
#pragma unroll
  for (int m = 16; m > 0; m >>= 1) ss += __shfl_xor(ss, m, 32);
  const float inv = rsqrtf(ss * (1.0f / D_) + 1e-6f);
#pragma unroll
  for (int i = 0; i < 8; ++i) {
    const int c = i * 32 + lane;
    const float y = vals[i] * inv * w[c];
    if (Of) Of[(size_t)row * D_ + c] = y;
    if (Ob) Ob[(size_t)row * D_ + c] = (bf16)y;
  }
}

// ---------------------------------------------------------------------------
// Sampling locations + per-(b,n,h) softmax of attention weights
// ---------------------------------------------------------------------------
__global__ void k_locaw(const float* __restrict__ off, const float* __restrict__ aw,
                        const float* __restrict__ refp, float* __restrict__ loc,
                        float* __restrict__ awsm) {
  int i = blockIdx.x * 256 + threadIdx.x;  // over M_*HEADS_
  if (i >= M_ * HEADS_) return;
  const int h = i & 7;
  const int t = i >> 3;
  const float cx = refp[t * 4 + 0], cy = refp[t * 4 + 1];
  const float sx = refp[t * 4 + 2] * 0.5f * (1.0f / NP_);
  const float sy = refp[t * 4 + 3] * 0.5f * (1.0f / NP_);
  const float* o = off + (size_t)t * 256 + h * 32;   // (h*TP + p)*2 + c
  const float* a = aw  + (size_t)t * 128 + h * 16;
  float mx = -3.4e38f;
  for (int p = 0; p < TP_; ++p) mx = fmaxf(mx, a[p]);
  float e[TP_], s = 0.f;
  for (int p = 0; p < TP_; ++p) { e[p] = __expf(a[p] - mx); s += e[p]; }
  const float inv = 1.0f / s;
  for (int p = 0; p < TP_; ++p) {
    loc[((size_t)i) * 32 + p * 2 + 0] = cx + o[p * 2 + 0] * sx;
    loc[((size_t)i) * 32 + p * 2 + 1] = cy + o[p * 2 + 1] * sy;
    awsm[((size_t)i) * 16 + p] = e[p] * inv;
  }
}

// ---------------------------------------------------------------------------
// Deformable bilinear sampling: wave per (b,n,h), lane = channel (HD=32).
// Gathers are clamped in-bounds; out-of-range corners contribute zero weight.
// ---------------------------------------------------------------------------
__global__ __launch_bounds__(256)
void k_sample(const float* __restrict__ memory, const float* __restrict__ loc,
              const float* __restrict__ awsm, float* __restrict__ ca) {
  const int lane = threadIdx.x & 31;
  const int wid  = threadIdx.x >> 5;
  const int idx  = blockIdx.x * 8 + wid;  // over M_*HEADS_
  const int h = idx & 7;
  const int t = idx >> 3;
  const int b = t / N_;
  const float* mem = memory + ((size_t)b * L_) * D_ + h * HD_ + lane;
  const float* lc = loc + (size_t)idx * 32;
  const float* wv = awsm + (size_t)idx * 16;
  const int Hs[4] = {80, 40, 20, 10};
  const int Ws[4] = {80, 40, 20, 10};
  const int St[4] = {0, 6400, 8000, 8400};
  float acc = 0.f;
#pragma unroll
  for (int p = 0; p < TP_; ++p) {
    const int li = p >> 2;
    const int Hh = Hs[li], Ww = Ws[li], st = St[li];
    const float x = lc[p * 2 + 0] * Ww - 0.5f;
    const float y = lc[p * 2 + 1] * Hh - 0.5f;
    const float x0f = floorf(x), y0f = floorf(y);
    const float dx = x - x0f, dy = y - y0f;
    const int x0 = (int)x0f, y0 = (int)y0f;
    const float wgt = wv[p];
    auto g = [&](int xi, int yi, float w) -> float {
      const bool valid = (xi >= 0) && (xi < Ww) && (yi >= 0) && (yi < Hh);
      int cx = xi < 0 ? 0 : (xi > Ww - 1 ? Ww - 1 : xi);
      int cy = yi < 0 ? 0 : (yi > Hh - 1 ? Hh - 1 : yi);
      const float val = mem[(size_t)(st + cy * Ww + cx) * D_];
      return valid ? val * w : 0.0f;
    };
    float s = g(x0, y0, (1.f - dx) * (1.f - dy)) + g(x0 + 1, y0, dx * (1.f - dy)) +
              g(x0, y0 + 1, (1.f - dx) * dy)     + g(x0 + 1, y0 + 1, dx * dy);
    acc += wgt * s;
  }
  ca[(size_t)t * D_ + h * HD_ + lane] = acc;
}

// ---------------------------------------------------------------------------
// SwiGLU: hp = silu(h1) * h2 (bf16 in, bf16 out for the w2 WMMA GEMM)
// ---------------------------------------------------------------------------
__global__ void k_silu(const bf16* __restrict__ hb, bf16* __restrict__ hp) {
  int i = blockIdx.x * 256 + threadIdx.x;  // over M_*HID_
  if (i >= M_ * HID_) return;
  const int t = i / HID_;
  const int j = i - t * HID_;
  const float h1 = (float)hb[(size_t)t * FF2_ + j];
  const float h2 = (float)hb[(size_t)t * FF2_ + HID_ + j];
  const float sg = h1 / (1.0f + __expf(-h1));
  hp[i] = (bf16)(sg * h2);
}

// ---------------------------------------------------------------------------
extern "C" void kernel_launch(void* const* d_in, const int* in_sizes, int n_in,
                              void* d_out, int out_size, void* d_ws, size_t ws_size,
                              hipStream_t stream) {
  (void)in_sizes; (void)n_in; (void)out_size; (void)ws_size;
  const float* target = (const float*)d_in[0];
  const float* refp   = (const float*)d_in[1];
  const float* memory = (const float*)d_in[2];
  const float* qkv_w  = (const float*)d_in[3];
  const float* qkv_b  = (const float*)d_in[4];
  const float* out_w  = (const float*)d_in[5];
  const float* out_b  = (const float*)d_in[6];
  const float* n1w    = (const float*)d_in[7];
  const float* off_w  = (const float*)d_in[8];
  const float* off_b  = (const float*)d_in[9];
  const float* aw_w   = (const float*)d_in[10];
  const float* aw_b   = (const float*)d_in[11];
  const float* cout_w = (const float*)d_in[12];
  const float* cout_b = (const float*)d_in[13];
  const float* n2w    = (const float*)d_in[14];
  const float* w13    = (const float*)d_in[15];
  const float* w13_b  = (const float*)d_in[16];
  const float* w2     = (const float*)d_in[17];
  const float* w2_b   = (const float*)d_in[18];
  const float* n3w    = (const float*)d_in[19];

  char* base = (char*)d_ws;
  size_t off = 0;
  auto alloc = [&](size_t bytes) -> char* {
    char* p = base + off;
    off = (off + bytes + 255) & ~(size_t)255;
    return p;
  };

  // bf16 weights
  bf16* wb_qkv  = (bf16*)alloc((size_t)768 * 256 * 2);
  bf16* wb_out  = (bf16*)alloc((size_t)256 * 256 * 2);
  bf16* wb_off  = (bf16*)alloc((size_t)256 * 256 * 2);
  bf16* wb_aw   = (bf16*)alloc((size_t)128 * 256 * 2);
  bf16* wb_cout = (bf16*)alloc((size_t)256 * 256 * 2);
  bf16* wb_w13  = (bf16*)alloc((size_t)FF2_ * 256 * 2);
  bf16* wb_w2   = (bf16*)alloc((size_t)256 * HID_ * 2);
  // activations / intermediates
  bf16*  tb   = (bf16*)alloc((size_t)M_ * D_ * 2);
  float* qkv  = (float*)alloc((size_t)M_ * 768 * 4);
  bf16*  Qb   = (bf16*)alloc((size_t)B_ * HEADS_ * NPAD_ * HD_ * 2);
  bf16*  Kb   = (bf16*)alloc((size_t)B_ * HEADS_ * NPAD_ * HD_ * 2);
  bf16*  Vtb  = (bf16*)alloc((size_t)B_ * HEADS_ * HD_ * NPAD_ * 2);
  float* sa   = (float*)alloc((size_t)M_ * D_ * 4);
  bf16*  sab  = (bf16*)alloc((size_t)M_ * D_ * 2);
  float* sao  = (float*)alloc((size_t)M_ * D_ * 4);
  float* x1   = (float*)alloc((size_t)M_ * D_ * 4);
  bf16*  x1b  = (bf16*)alloc((size_t)M_ * D_ * 2);
  float* offs = (float*)alloc((size_t)M_ * 256 * 4);
  float* aw   = (float*)alloc((size_t)M_ * 128 * 4);
  float* loc  = (float*)alloc((size_t)M_ * HEADS_ * TP_ * 2 * 4);
  float* awsm = (float*)alloc((size_t)M_ * HEADS_ * TP_ * 4);
  float* ca   = (float*)alloc((size_t)M_ * D_ * 4);
  bf16*  cab  = (bf16*)alloc((size_t)M_ * D_ * 2);
  float* cao  = (float*)alloc((size_t)M_ * D_ * 4);
  float* x2   = (float*)alloc((size_t)M_ * D_ * 4);
  bf16*  x2b  = (bf16*)alloc((size_t)M_ * D_ * 2);
  bf16*  hb   = (bf16*)alloc((size_t)M_ * FF2_ * 2);
  bf16*  hp   = (bf16*)alloc((size_t)M_ * HID_ * 2);
  float* ffn  = (float*)alloc((size_t)M_ * D_ * 4);

  auto cvt = [&](const float* s, bf16* d, int n) {
    k_cvt<<<(n + 255) / 256, 256, 0, stream>>>(s, d, n);
  };
  // weight conversion (cheap: ~0.8M elements)
  cvt(qkv_w,  wb_qkv,  768 * 256);
  cvt(out_w,  wb_out,  256 * 256);
  cvt(off_w,  wb_off,  256 * 256);
  cvt(aw_w,   wb_aw,   128 * 256);
  cvt(cout_w, wb_cout, 256 * 256);
  cvt(w13,    wb_w13,  FF2_ * 256);
  cvt(w2,     wb_w2,   256 * HID_);
  cvt(target, tb, M_ * D_);

  const dim3 blk32(32), blk256(256);

  // qkv projection: (M x 256) x (768 x 256)^T
  k_gemm<<<dim3(M_ / BM_, 768 / BN_), blk256, 0, stream>>>(tb, wb_qkv, qkv_b, qkv,
                                                           nullptr, M_, 768, D_);
  // repack into padded Q/K/Vt
  k_repack<<<(B_ * HEADS_ * NPAD_ * HD_ + 255) / 256, blk256, 0, stream>>>(qkv, Qb, Kb, Vtb);
  // attention (128 bh x 19 q-tiles)
  k_attn<<<dim3(B_ * HEADS_, 19), blk32, 0, stream>>>(Qb, Kb, Vtb, sa);
  // out projection + residual + rmsnorm1
  cvt(sa, sab, M_ * D_);
  k_gemm<<<dim3(M_ / BM_, D_ / BN_), blk256, 0, stream>>>(sab, wb_out, out_b, sao,
                                                          nullptr, M_, D_, D_);
  k_addnorm<<<M_ / 8, blk256, 0, stream>>>(target, sao, n1w, x1, x1b, 0);

  // deformable cross-attention
  k_gemm<<<dim3(M_ / BM_, 256 / BN_), blk256, 0, stream>>>(x1b, wb_off, off_b, offs,
                                                           nullptr, M_, 256, D_);
  k_gemm<<<dim3(M_ / BM_, 128 / BN_), blk256, 0, stream>>>(x1b, wb_aw, aw_b, aw,
                                                           nullptr, M_, 128, D_);
  k_locaw<<<(M_ * HEADS_ + 255) / 256, blk256, 0, stream>>>(offs, aw, refp, loc, awsm);
  k_sample<<<(M_ * HEADS_) / 8, blk256, 0, stream>>>(memory, loc, awsm, ca);
  cvt(ca, cab, M_ * D_);
  k_gemm<<<dim3(M_ / BM_, D_ / BN_), blk256, 0, stream>>>(cab, wb_cout, cout_b, cao,
                                                          nullptr, M_, D_, D_);
  k_addnorm<<<M_ / 8, blk256, 0, stream>>>(x1, cao, n2w, x2, x2b, 0);

  // SwiGLU FFN
  k_gemm<<<dim3(M_ / BM_, FF2_ / BN_), blk256, 0, stream>>>(x2b, wb_w13, w13_b, nullptr,
                                                            hb, M_, FF2_, D_);
  k_silu<<<(M_ * HID_ + 255) / 256, blk256, 0, stream>>>(hb, hp);
  k_gemm<<<dim3(M_ / BM_, D_ / BN_), blk256, 0, stream>>>(hp, wb_w2, w2_b, ffn,
                                                          nullptr, M_, D_, HID_);
  // residual + clip + final rmsnorm -> d_out (f32)
  k_addnorm<<<M_ / 8, blk256, 0, stream>>>(x2, ffn, n3w, (float*)d_out, nullptr, 1);
}
